// RNNDecoder_25864293057254
// MI455X (gfx1250) — matrix operand order repaired and checked by reading, not compiled, not run on hardware
//
#include <hip/hip_runtime.h>
#include <hip/hip_bf16.h>

// ---------------------------------------------------------------------------
// Types for CDNA5 WMMA (wave32, 16x16x32 bf16 -> f32)
// ---------------------------------------------------------------------------
typedef __attribute__((ext_vector_type(16))) __bf16 v16bf;
typedef __attribute__((ext_vector_type(8)))  float  v8f;
typedef __attribute__((ext_vector_type(4)))  unsigned v4u_t;
typedef __attribute__((ext_vector_type(8)))  int      v8i_t;
typedef __attribute__((ext_vector_type(4)))  int      v4i_t;
typedef unsigned short u16;

union ABu { v16bf v; u16 u[16]; };

#define DEV __device__ __forceinline__

// Problem constants
#define BB    16
#define TT    241
#define TP    256
#define HH    1024
#define G4    4096
#define NC    100
#define NCP   128
#define TE    512
#define NHEAD 4
#define DHEAD 256

// Padded LDS A-tile layout: rows of 1024 bf16, +16B pad every 512 elements
// (TDM pad_interval=256 DWORDs, pad_amount=4 DWORDs) -> row stride 2080 B.
#define A_ROW_BYTES 2080
#define A_TILE_BYTES (16 * A_ROW_BYTES)   // 33280

#if __has_builtin(__builtin_amdgcn_tensor_load_to_lds) && \
    __has_builtin(__builtin_amdgcn_s_wait_tensorcnt)
#define USE_TDM 1
#else
#define USE_TDM 0
#endif

DEV u16 f2bf(float f) {
    unsigned u = __builtin_bit_cast(unsigned, f);
    unsigned r = (u + 0x7FFFu + ((u >> 16) & 1u)) >> 16;
    return (u16)r;
}
DEV float sigm(float x) { return 1.0f / (1.0f + expf(-x)); }

// A fragment (16x32 bf16) from row-major global memory, leading dim lda.
DEV v16bf load_a_frag(const u16* A, int lda, int k0, int lane) {
    int m  = lane & 15;
    int kb = k0 + ((lane >> 4) << 3);
    const u16* p = A + (long)m * lda + kb;
    ABu f;
#pragma unroll
    for (int j = 0; j < 8; ++j) { f.u[j] = p[j]; f.u[j + 8] = p[16 + j]; }
    return f.v;
}

// A fragment from the padded LDS tile (1024-element rows).
DEV v16bf load_a_frag_lds(const u16* smem, int k0, int lane) {
    int m  = lane & 15;
    int kb = k0 + ((lane >> 4) << 3);
    int kb2 = kb + 16;
    const char* base = (const char*)smem + (long)m * A_ROW_BYTES;
    const u16* p0 = (const u16*)(base + kb * 2 + ((kb >> 9) << 4));
    const u16* p1 = (const u16*)(base + kb2 * 2 + ((kb2 >> 9) << 4));
    ABu f;
#pragma unroll
    for (int j = 0; j < 8; ++j) { f.u[j] = p0[j]; f.u[j + 8] = p1[j]; }
    return f.v;
}

// B fragment: 32x16 bf16 tile from (K x N) row-major packed matrix.
DEV v16bf load_b_frag(const u16* B, int ldb, int k0, int n0, int lane) {
    const u16* p = B + (long)(k0 + ((lane >> 4) << 3)) * ldb + n0 + (lane & 15);
    ABu f;
#pragma unroll
    for (int j = 0; j < 8; ++j) { f.u[j] = p[(long)j * ldb]; f.u[j + 8] = p[(long)(16 + j) * ldb]; }
    return f.v;
}

DEV v8f wmma_bf16(v16bf a, v16bf b, v8f c) {
    return __builtin_amdgcn_wmma_f32_16x16x32_bf16(false, a, false, b,
                                                   (short)0, c, false, false);
}

// ---------------------------------------------------------------------------
// Stage a 16 x 1024 bf16 A-tile (row stride lda elements) into LDS offset 0
// via the Tensor Data Mover (TDM); cooperative-copy fallback otherwise.
// Caller must __syncthreads() afterwards. Kernel must have no static LDS so
// the dynamic-LDS block sits at offset 0.
// ---------------------------------------------------------------------------
DEV void stage_a_tile(u16* smem, const u16* g, int lda) {
#if USE_TDM
    if (threadIdx.x < 32) {
        union { v4u_t v; unsigned u[4]; } G0;
        union { v8i_t v; unsigned u[8]; } G1;
        union { v4i_t v; int i[4]; } Z4;
        union { v8i_t v; int i[8]; } Z8;
        unsigned long long ga = (unsigned long long)(const void*)g;
        G0.u[0] = 1u;                                            // count=1
        G0.u[1] = 0u;                                            // lds_addr
        G0.u[2] = (unsigned)ga;                                  // gaddr lo
        G0.u[3] = (unsigned)((ga >> 32) & 0x01FFFFFFu) | 0x80000000u; // type=2
        // data_size=2B, pad_enable, pad_interval=256 DW, pad_amount=4 DW
        G1.u[0] = (1u << 16) | (1u << 20) | (7u << 22) | (3u << 25);
        G1.u[1] = ((unsigned)lda & 0xFFFFu) << 16;               // tensor_dim0 lo
        G1.u[2] = ((unsigned)lda >> 16) | (16u << 16);           // td0 hi|td1 lo
        G1.u[3] = (1024u << 16);                                 // tile_dim0
        G1.u[4] = 16u;                                           // tile_dim1
        G1.u[5] = (unsigned)lda;                                 // dim0 stride lo
        G1.u[6] = 0u;
        G1.u[7] = 0u;
        Z4.i[0] = Z4.i[1] = Z4.i[2] = Z4.i[3] = 0;
#pragma unroll
        for (int j = 0; j < 8; ++j) Z8.i[j] = 0;
        __builtin_amdgcn_tensor_load_to_lds(G0.v, G1.v, Z4.v, Z4.v, Z8.v, 0);
        __builtin_amdgcn_s_wait_tensorcnt((short)0);
    }
#else
    for (int e = threadIdx.x; e < 16 * 1024; e += (int)blockDim.x) {
        int m = e >> 10, k = e & 1023;
        *(u16*)((char*)smem + (long)m * A_ROW_BYTES + k * 2 + ((k >> 9) << 4)) =
            g[(long)m * lda + k];
    }
#endif
}

// ---------------------------------------------------------------------------
// Grid-wide split-phase barrier (monotonic counter, 8 co-resident blocks)
// ---------------------------------------------------------------------------
DEV void gsync(unsigned* counter, unsigned nb) {
    __syncthreads();
    if (threadIdx.x == 0) {
        __threadfence();
        unsigned arrive = __hip_atomic_fetch_add(counter, 1u, __ATOMIC_ACQ_REL,
                                                 __HIP_MEMORY_SCOPE_AGENT);
        unsigned target = (arrive / nb + 1u) * nb;
        while (__hip_atomic_load(counter, __ATOMIC_ACQUIRE,
                                 __HIP_MEMORY_SCOPE_AGENT) < target) {
            __builtin_amdgcn_s_sleep(1);
        }
    }
    __syncthreads();
    __threadfence();
}

// ---------------------------------------------------------------------------
// Small prep kernels
// ---------------------------------------------------------------------------
__global__ void zero_u32_k(unsigned* p, long n) {
    for (long i = (long)blockIdx.x * blockDim.x + threadIdx.x; i < n;
         i += (long)gridDim.x * blockDim.x) p[i] = 0u;
}

__global__ void cast_bf16_k(u16* dst, const float* src, long n) {
    for (long i = (long)blockIdx.x * blockDim.x + threadIdx.x; i < n;
         i += (long)gridDim.x * blockDim.x) dst[i] = f2bf(src[i]);
}

// dst (K x N) row-major <- src (N x K) row-major  (packs W so x@W^T is A@B)
__global__ void transpose_cast_k(u16* dst, const float* src, int N, int K) {
    long tot = (long)N * K;
    for (long i = (long)blockIdx.x * blockDim.x + threadIdx.x; i < tot;
         i += (long)gridDim.x * blockDim.x) {
        long k = i / N, n = i % N;
        dst[i] = f2bf(src[n * (long)K + k]);
    }
}

__global__ void pad_w2_k(u16* dst, const float* src) {   // 1024x100 -> 1024x128
    long tot = (long)HH * NCP;
    for (long i = (long)blockIdx.x * blockDim.x + threadIdx.x; i < tot;
         i += (long)gridDim.x * blockDim.x) {
        long k = i / NCP, n = i % NCP;
        dst[i] = (n < NC) ? f2bf(src[k * NC + n]) : (u16)0;
    }
}

__global__ void pad_b2_k(float* dst, const float* src) {
    int i = threadIdx.x;
    if (i < NCP) dst[i] = (i < NC) ? src[i] : 0.0f;
}

__global__ void bias_sum_k(float* dst, const float* a, const float* b, int n) {
    for (int i = blockIdx.x * blockDim.x + threadIdx.x; i < n;
         i += gridDim.x * blockDim.x) dst[i] = a[i] + b[i];
}

__global__ void embed_k(float* ef, u16* eb, const int* tok, const float* table) {
    long tot = (long)BB * TP * HH;
    for (long i = (long)blockIdx.x * blockDim.x + threadIdx.x; i < tot;
         i += (long)gridDim.x * blockDim.x) {
        long row = i >> 10; int d = (int)(i & 1023);
        int b = (int)(row >> 8), t = (int)(row & 255);
        float v = 0.0f;
        if (t < TT) v = table[(long)tok[b * TT + t] * HH + d];
        ef[i] = v; eb[i] = f2bf(v);
    }
}

// ---------------------------------------------------------------------------
// Generic bf16 WMMA GEMM with TDM-staged A tile in LDS and ping-pong
// double-buffered fragment loads (no inter-buffer register copies).
// C(MxN,f32) = A(MxK)*B(KxN)+bias, opt tanh, optional f32/bf16 outputs
// (ldc == N). M%16==0, K%1024==0, N%64==0, lda==K.
// grid = (ceil(N/512), M/16), block = 256 (8 waves, 16x64 tile each),
// dynamic LDS = A_TILE_BYTES.
// ---------------------------------------------------------------------------
__global__ void gemm_bf16_k(const u16* __restrict__ A, int lda,
                            const u16* __restrict__ Bm, int ldb,
                            const float* __restrict__ bias,
                            float* __restrict__ Cf, u16* __restrict__ Cb,
                            int M, int N, int K, int act) {
    extern __shared__ u16 sA[];
    int wave = threadIdx.x >> 5, lane = threadIdx.x & 31;
    int n0 = (blockIdx.x * 8 + wave) * 64;
    int m0 = blockIdx.y * 16;
    bool active = n0 < N;
    v8f z = {0.f,0.f,0.f,0.f,0.f,0.f,0.f,0.f};
    v8f acc[4] = {z, z, z, z};

    for (int kc = 0; kc < K; kc += 1024) {
        stage_a_tile(sA, A + (long)m0 * lda + kc, lda);
        __syncthreads();
        if (active) {
            v16bf a0 = load_a_frag_lds(sA, 0, lane);
            v16bf a1 = load_a_frag_lds(sA, 32, lane);
            v16bf b0[4], b1[4];
#pragma unroll
            for (int t = 0; t < 4; ++t) {
                b0[t] = load_b_frag(Bm, ldb, kc, n0 + 16 * t, lane);
                b1[t] = load_b_frag(Bm, ldb, kc + 32, n0 + 16 * t, lane);
            }
            for (int k = 0; k < 1024; k += 64) {
                __builtin_prefetch(Bm + (long)(kc + k + 64 + lane) * ldb + n0, 0, 1);
#pragma unroll
                for (int t = 0; t < 4; ++t)
                    acc[t] = wmma_bf16(a0, b0[t], acc[t]);
                if (k + 64 < 1024) {
                    a0 = load_a_frag_lds(sA, k + 64, lane);
#pragma unroll
                    for (int t = 0; t < 4; ++t)
                        b0[t] = load_b_frag(Bm, ldb, kc + k + 64, n0 + 16 * t, lane);
                }
#pragma unroll
                for (int t = 0; t < 4; ++t)
                    acc[t] = wmma_bf16(a1, b1[t], acc[t]);
                if (k + 96 < 1024) {
                    a1 = load_a_frag_lds(sA, k + 96, lane);
#pragma unroll
                    for (int t = 0; t < 4; ++t)
                        b1[t] = load_b_frag(Bm, ldb, kc + k + 96, n0 + 16 * t, lane);
                }
            }
        }
        __syncthreads();
    }

    if (active) {
#pragma unroll
        for (int t = 0; t < 4; ++t) {
            int n = n0 + 16 * t + (lane & 15);
            float bv = bias ? bias[n] : 0.0f;
#pragma unroll
            for (int r = 0; r < 8; ++r) {
                int m = m0 + r + ((lane >> 4) << 3);
                float v = acc[t][r] + bv;
                if (act == 1) v = tanhf(v);
                long off = (long)m * N + n;
                if (Cf) Cf[off] = v;
                if (Cb) Cb[off] = f2bf(v);
            }
        }
    }
}

// ---------------------------------------------------------------------------
// Persistent LSTM scan. Per step: TDM-stage hprev (16x1024 bf16) into LDS,
// g = hprev @ Whh^T + xg[:,t,:] via WMMA (ping-pong buffers), grid-sync,
// elementwise cell, grid-sync. 8 blocks x 256 threads; dynamic LDS.
// ---------------------------------------------------------------------------
__global__ void lstm_scan_k(const float* __restrict__ xg,
                            const u16* __restrict__ whhT,
                            float* __restrict__ hsf, u16* __restrict__ hsb,
                            u16* __restrict__ hprev, float* __restrict__ cst,
                            float* __restrict__ gbuf, unsigned* counter) {
    extern __shared__ u16 sA[];
    const unsigned NB = 8;
    int tid = blockIdx.x * 256 + threadIdx.x;        // 0..2047
    // init: zero recurrent state and the padded tail rows t in [241,256)
    for (int i = tid; i < BB * HH; i += 2048) { hprev[i] = 0; cst[i] = 0.0f; }
    for (long i = tid; i < (long)BB * (TP - TT) * HH; i += 2048) {
        int b = (int)(i / ((TP - TT) * HH));
        int r = (int)((i / HH) % (TP - TT));
        int d = (int)(i & (HH - 1));
        long row = (long)b * TP + TT + r;
        if (hsf) hsf[row * HH + d] = 0.0f;
        hsb[row * HH + d] = 0;
    }
    gsync(counter, NB);

    int wave = threadIdx.x >> 5, lane = threadIdx.x & 31;
    int n0 = (blockIdx.x * 8 + wave) * 64;
    v8f z = {0.f,0.f,0.f,0.f,0.f,0.f,0.f,0.f};

    for (int t = 0; t < TT; ++t) {
        // phase A: gate pre-activations
        stage_a_tile(sA, hprev, HH);
        __syncthreads();
        v8f acc[4] = {z, z, z, z};
        {
            v16bf a0 = load_a_frag_lds(sA, 0, lane);
            v16bf a1 = load_a_frag_lds(sA, 32, lane);
            v16bf b0[4], b1[4];
#pragma unroll
            for (int q = 0; q < 4; ++q) {
                b0[q] = load_b_frag(whhT, G4, 0, n0 + 16 * q, lane);
                b1[q] = load_b_frag(whhT, G4, 32, n0 + 16 * q, lane);
            }
            for (int k = 0; k < HH; k += 64) {
#pragma unroll
                for (int q = 0; q < 4; ++q)
                    acc[q] = wmma_bf16(a0, b0[q], acc[q]);
                if (k + 64 < HH) {
                    a0 = load_a_frag_lds(sA, k + 64, lane);
#pragma unroll
                    for (int q = 0; q < 4; ++q)
                        b0[q] = load_b_frag(whhT, G4, k + 64, n0 + 16 * q, lane);
                }
#pragma unroll
                for (int q = 0; q < 4; ++q)
                    acc[q] = wmma_bf16(a1, b1[q], acc[q]);
                if (k + 96 < HH) {
                    a1 = load_a_frag_lds(sA, k + 96, lane);
#pragma unroll
                    for (int q = 0; q < 4; ++q)
                        b1[q] = load_b_frag(whhT, G4, k + 96, n0 + 16 * q, lane);
                }
            }
        }
#pragma unroll
        for (int q = 0; q < 4; ++q) {
#pragma unroll
            for (int r = 0; r < 8; ++r) {
                int m = r + ((lane >> 4) << 3);           // batch index
                int n = n0 + 16 * q + (lane & 15);
                gbuf[m * G4 + n] = acc[q][r] + xg[((long)m * TP + t) * G4 + n];
            }
        }
        gsync(counter, NB);
        // phase B: LSTM cell (gate order i,f,g,o)
        for (int e = tid; e < BB * HH; e += 2048) {
            int b = e >> 10, j = e & (HH - 1);
            float ig = gbuf[b * G4 + j];
            float fg = gbuf[b * G4 + HH + j];
            float gg = gbuf[b * G4 + 2 * HH + j];
            float og = gbuf[b * G4 + 3 * HH + j];
            float cc = sigm(fg) * cst[e] + sigm(ig) * tanhf(gg);
            float hh = sigm(og) * tanhf(cc);
            cst[e] = cc;
            long row = (long)b * TP + t;
            if (hsf) hsf[row * HH + j] = hh;
            u16 hb = f2bf(hh);
            hsb[row * HH + j] = hb;
            hprev[e] = hb;
        }
        gsync(counter, NB);
    }
}

// K^T pack for attention: (B*TE, H) -> (b, h, d, tk)
__global__ void kT_pack_k(u16* kT, const u16* kp) {
    long tot = (long)BB * NHEAD * DHEAD * TE;
    for (long i = (long)blockIdx.x * blockDim.x + threadIdx.x; i < tot;
         i += (long)gridDim.x * blockDim.x) {
        long tmp = i;
        int t = (int)(tmp % TE); tmp /= TE;
        int d = (int)(tmp % DHEAD); tmp /= DHEAD;
        int h = (int)(tmp % NHEAD);
        int b = (int)(tmp / NHEAD);
        kT[i] = kp[((long)b * TE + t) * HH + h * DHEAD + d];
    }
}

// ---------------------------------------------------------------------------
// Attention: one block per (qt, head, b).  scores = q k^T / 16 (WMMA),
// f32 softmax in LDS, probs -> bf16 LDS, ctx = P V (WMMA). Ping-pong
// double-buffered fragment loads in both WMMA phases.
// ---------------------------------------------------------------------------
__global__ void attn_k(const u16* __restrict__ qb, const u16* __restrict__ kT,
                       const u16* __restrict__ vb, float* __restrict__ attn_out,
                       float* __restrict__ ctx) {
    __shared__ float sc[16][TE];
    __shared__ u16   pb[16][TE];
    __shared__ float red[256];
    __shared__ float rowred[2][16];

    int qt = blockIdx.x, h = blockIdx.y, b = blockIdx.z;
    int wave = threadIdx.x >> 5, lane = threadIdx.x & 31;
    v8f z = {0.f,0.f,0.f,0.f,0.f,0.f,0.f,0.f};

    // phase 1: scores (each wave: 16 x 64 keys)
    {
        const u16* Aq = qb + ((long)b * TP + qt * 16) * HH + h * DHEAD;
        const u16* Bk = kT + ((long)b * NHEAD + h) * DHEAD * TE;
        int n0 = wave * 64;
        v8f acc[4] = {z, z, z, z};
        v16bf a0 = load_a_frag(Aq, HH, 0, lane);
        v16bf a1 = load_a_frag(Aq, HH, 32, lane);
        v16bf b0[4], b1[4];
#pragma unroll
        for (int q = 0; q < 4; ++q) {
            b0[q] = load_b_frag(Bk, TE, 0, n0 + 16 * q, lane);
            b1[q] = load_b_frag(Bk, TE, 32, n0 + 16 * q, lane);
        }
        for (int k = 0; k < DHEAD; k += 64) {
#pragma unroll
            for (int q = 0; q < 4; ++q)
                acc[q] = wmma_bf16(a0, b0[q], acc[q]);
            if (k + 64 < DHEAD) {
                a0 = load_a_frag(Aq, HH, k + 64, lane);
#pragma unroll
                for (int q = 0; q < 4; ++q)
                    b0[q] = load_b_frag(Bk, TE, k + 64, n0 + 16 * q, lane);
            }
#pragma unroll
            for (int q = 0; q < 4; ++q)
                acc[q] = wmma_bf16(a1, b1[q], acc[q]);
            if (k + 96 < DHEAD) {
                a1 = load_a_frag(Aq, HH, k + 96, lane);
#pragma unroll
                for (int q = 0; q < 4; ++q)
                    b1[q] = load_b_frag(Bk, TE, k + 96, n0 + 16 * q, lane);
            }
        }
#pragma unroll
        for (int q = 0; q < 4; ++q) {
#pragma unroll
            for (int r = 0; r < 8; ++r) {
                int m = r + ((lane >> 4) << 3);
                int n = n0 + 16 * q + (lane & 15);
                sc[m][n] = acc[q][r] * 0.0625f;     // 1/sqrt(256)
            }
        }
    }
    __syncthreads();

    // phase 2: softmax over 512 keys per row (16 partials x 32 cols)
    {
        int row = threadIdx.x & 15, part = threadIdx.x >> 4;
        int c0 = part * 32;
        float pm = -1e30f;
        for (int c = c0; c < c0 + 32; ++c) pm = fmaxf(pm, sc[row][c]);
        red[threadIdx.x] = pm; __syncthreads();
        if (part == 0) {
            float m = red[row];
            for (int p = 1; p < 16; ++p) m = fmaxf(m, red[p * 16 + row]);
            rowred[0][row] = m;
        }
        __syncthreads();
        float rmax = rowred[0][row];
        float ps = 0.0f;
        for (int c = c0; c < c0 + 32; ++c) ps += expf(sc[row][c] - rmax);
        red[threadIdx.x] = ps; __syncthreads();
        if (part == 0) {
            float s = red[row];
            for (int p = 1; p < 16; ++p) s += red[p * 16 + row];
            rowred[1][row] = 1.0f / s;
        }
        __syncthreads();
        float inv = rowred[1][row];
        int qrow = qt * 16 + row;
        long aoff = (((long)b * NHEAD + h) * TT + qrow) * TE;
        for (int c = c0; c < c0 + 32; ++c) {
            float p = expf(sc[row][c] - rmax) * inv;
            pb[row][c] = f2bf(p);
            if (qrow < TT) attn_out[aoff + c] = p;
        }
    }
    __syncthreads();

    // phase 3: ctx = P @ V  (each wave: 16 x 32 of dh)
    {
        const u16* Bv = vb + (long)b * TE * HH + h * DHEAD;
        const u16* Ap = (const u16*)&pb[0][0];
        int n0 = wave * 32;
        v8f acc[2] = {z, z};
        v16bf a0 = load_a_frag(Ap, TE, 0, lane);
        v16bf a1 = load_a_frag(Ap, TE, 32, lane);
        v16bf b0[2], b1[2];
#pragma unroll
        for (int q = 0; q < 2; ++q) {
            b0[q] = load_b_frag(Bv, HH, 0, n0 + 16 * q, lane);
            b1[q] = load_b_frag(Bv, HH, 32, n0 + 16 * q, lane);
        }
        for (int k = 0; k < TE; k += 64) {
#pragma unroll
            for (int q = 0; q < 2; ++q)
                acc[q] = wmma_bf16(a0, b0[q], acc[q]);
            if (k + 64 < TE) {
                a0 = load_a_frag(Ap, TE, k + 64, lane);
#pragma unroll
                for (int q = 0; q < 2; ++q)
                    b0[q] = load_b_frag(Bv, HH, k + 64, n0 + 16 * q, lane);
            }
#pragma unroll
            for (int q = 0; q < 2; ++q)
                acc[q] = wmma_bf16(a1, b1[q], acc[q]);
            if (k + 96 < TE) {
                a1 = load_a_frag(Ap, TE, k + 96, lane);
#pragma unroll
                for (int q = 0; q < 2; ++q)
                    b1[q] = load_b_frag(Bv, HH, k + 96, n0 + 16 * q, lane);
            }
        }
#pragma unroll
        for (int q = 0; q < 2; ++q) {
#pragma unroll
            for (int r = 0; r < 8; ++r) {
                int m = r + ((lane >> 4) << 3);
                int n = n0 + 16 * q + (lane & 15);
                ctx[((long)b * TP + qt * 16 + m) * HH + h * DHEAD + n] = acc[q][r];
            }
        }
    }
}

__global__ void feats_k(u16* f, const float* h1, const float* ctx) {
    long tot = (long)BB * TP * 2 * HH;
    for (long i = (long)blockIdx.x * blockDim.x + threadIdx.x; i < tot;
         i += (long)gridDim.x * blockDim.x) {
        long row = i >> 11; int c = (int)(i & 2047);
        int t = (int)(row & 255);
        float v = 0.0f;
        if (t < TT) v = (c < HH) ? h1[row * HH + c] : ctx[row * HH + (c - HH)];
        f[i] = f2bf(v);
    }
}

__global__ void logsoftmax_k(float* out, const float* logits) {
    int wave = threadIdx.x >> 5, lane = threadIdx.x & 31;
    long row = (long)blockIdx.x * 8 + wave;           // 0..3855
    if (row >= (long)BB * TT) return;
    int b = (int)(row / TT), t = (int)(row % TT);
    const float* lp = logits + ((long)b * TP + t) * NCP;
    float m = -1e30f;
    for (int c = lane; c < NC; c += 32) m = fmaxf(m, lp[c]);
    for (int o = 16; o; o >>= 1) m = fmaxf(m, __shfl_xor(m, o, 32));
    float s = 0.0f;
    for (int c = lane; c < NC; c += 32) s += expf(lp[c] - m);
    for (int o = 16; o; o >>= 1) s += __shfl_xor(s, o, 32);
    float lg = logf(s);
    for (int c = lane; c < NC; c += 32) out[row * NC + c] = lp[c] - m - lg;
}

// word-boundary scan (single thread; replicates _split_words_meta on tok[:,1:])
__global__ void words_meta_k(int* meta, const int* tok) {
    int* segB = meta + 4;
    int* segS = segB + 8192;
    int* segL = segS + 8192;
    int N = 0, maxlen = 0;
    for (int b = 0; b < BB; ++b) {
        int start = 0; bool broke = false;
        for (int t = 0; t < TT - 1; ++t) {
            int v = tok[b * TT + 1 + t];
            if (v == 1 || v == 2) {
                int l = t - start;
                segB[N] = b; segS[N] = start; segL[N] = l; ++N;
                if (l > maxlen) maxlen = l;
                start = t + 1;
                if (v == 2) { broke = true; break; }
            }
        }
        if (!broke && start < TT - 1) {
            int l = (TT - 1) - start;
            segB[N] = b; segS[N] = start; segL[N] = l; ++N;
            if (l > maxlen) maxlen = l;
        }
    }
    meta[0] = N; meta[1] = maxlen;
}

// words[n,l,:] = mask * concat(emb[b, pos+1], ctx[b, pos])
__global__ void words_gather_k(float* dst, const int* meta, const float* embf,
                               const float* ctx, long count) {
    const int* segB = meta + 4;
    const int* segS = segB + 8192;
    const int* segL = segS + 8192;
    int N = meta[0], ml = meta[1];
    long stride = (long)ml * 2048;
    for (long e = (long)blockIdx.x * blockDim.x + threadIdx.x; e < count;
         e += (long)gridDim.x * blockDim.x) {
        float v = 0.0f;
        if (stride > 0) {
            long n_ = e / stride;
            long rem = e % stride;
            int l = (int)(rem / 2048), d = (int)(rem & 2047);
            if (n_ < N && l < segL[n_]) {
                int b = segB[n_], pos = segS[n_] + l;
                v = (d < HH) ? embf[((long)b * TP + pos + 1) * HH + d]
                             : ctx[((long)b * TP + pos) * HH + (d - HH)];
            }
        }
        dst[e] = v;
    }
}

// ---------------------------------------------------------------------------
// Workspace layout (bytes, 256-aligned)
// ---------------------------------------------------------------------------
namespace wsl {
constexpr size_t AL(size_t x) { return (x + 255) & ~(size_t)255; }
constexpr size_t EMBF  = 0;
constexpr size_t EMBB  = AL(EMBF  + (size_t)BB*TP*HH*4);
constexpr size_t XG    = AL(EMBB  + (size_t)BB*TP*HH*2);
constexpr size_t H0B   = AL(XG    + (size_t)BB*TP*G4*4);
constexpr size_t H1F   = AL(H0B   + (size_t)BB*TP*HH*2);
constexpr size_t H1B   = AL(H1F   + (size_t)BB*TP*HH*4);
constexpr size_t QB    = AL(H1B   + (size_t)BB*TP*HH*2);
constexpr size_t ENCB  = AL(QB    + (size_t)BB*TP*HH*2);
constexpr size_t KPB   = AL(ENCB  + (size_t)BB*TE*HH*2);
constexpr size_t VB    = AL(KPB   + (size_t)BB*TE*HH*2);
constexpr size_t KTB   = AL(VB    + (size_t)BB*TE*HH*2);
constexpr size_t CTX   = AL(KTB   + (size_t)BB*TE*HH*2);
constexpr size_t FEATS = AL(CTX   + (size_t)BB*TP*HH*4);
constexpr size_t XB    = AL(FEATS + (size_t)BB*TP*2*HH*2);
constexpr size_t LOGIT = AL(XB    + (size_t)BB*TP*HH*2);
constexpr size_t WIT0  = AL(LOGIT + (size_t)BB*TP*NCP*4);
constexpr size_t WHT0  = AL(WIT0  + (size_t)HH*G4*2);
constexpr size_t WIT1  = AL(WHT0  + (size_t)HH*G4*2);
constexpr size_t WHT1  = AL(WIT1  + (size_t)HH*G4*2);
constexpr size_t WQB   = AL(WHT1  + (size_t)HH*G4*2);
constexpr size_t WKB   = AL(WQB   + (size_t)HH*HH*2);
constexpr size_t WVB   = AL(WKB   + (size_t)HH*HH*2);
constexpr size_t W1B   = AL(WVB   + (size_t)HH*HH*2);
constexpr size_t W2P   = AL(W1B   + (size_t)2*HH*HH*2);
constexpr size_t BS0   = AL(W2P   + (size_t)HH*NCP*2);
constexpr size_t BS1   = AL(BS0   + (size_t)G4*4);
constexpr size_t B2P   = AL(BS1   + (size_t)G4*4);
constexpr size_t GBUF  = AL(B2P   + (size_t)NCP*4);
constexpr size_t HPREV = AL(GBUF  + (size_t)BB*G4*4);
constexpr size_t CST   = AL(HPREV + (size_t)BB*HH*2);
constexpr size_t CNT   = AL(CST   + (size_t)BB*HH*4);
constexpr size_t META  = AL(CNT   + 2*4);
} // namespace wsl

extern "C" void kernel_launch(void* const* d_in, const int* in_sizes, int n_in,
                              void* d_out, int out_size, void* d_ws, size_t ws_size,
                              hipStream_t stream) {
    (void)in_sizes; (void)n_in; (void)ws_size;
    const int*   tok  = (const int*)  d_in[0];
    const float* enc  = (const float*)d_in[1];
    const float* emb  = (const float*)d_in[2];
    const float* wih0 = (const float*)d_in[3];
    const float* whh0 = (const float*)d_in[4];
    const float* bih0 = (const float*)d_in[5];
    const float* bhh0 = (const float*)d_in[6];
    const float* wih1 = (const float*)d_in[7];
    const float* whh1 = (const float*)d_in[8];
    const float* bih1 = (const float*)d_in[9];
    const float* bhh1 = (const float*)d_in[10];
    const float* wq = (const float*)d_in[11]; const float* bq = (const float*)d_in[12];
    const float* wk = (const float*)d_in[13]; const float* bk = (const float*)d_in[14];
    const float* wv = (const float*)d_in[15]; const float* bv = (const float*)d_in[16];
    const float* w1 = (const float*)d_in[17]; const float* b1 = (const float*)d_in[18];
    const float* w2 = (const float*)d_in[19]; const float* b2 = (const float*)d_in[20];

    char* ws = (char*)d_ws;
    float* embf  = (float*)(ws + wsl::EMBF);
    u16*   embb  = (u16*)  (ws + wsl::EMBB);
    float* xg    = (float*)(ws + wsl::XG);
    u16*   h0b   = (u16*)  (ws + wsl::H0B);
    float* h1f   = (float*)(ws + wsl::H1F);
    u16*   h1b   = (u16*)  (ws + wsl::H1B);
    u16*   qb    = (u16*)  (ws + wsl::QB);
    u16*   encb  = (u16*)  (ws + wsl::ENCB);
    u16*   kpb   = (u16*)  (ws + wsl::KPB);
    u16*   vbuf  = (u16*)  (ws + wsl::VB);
    u16*   ktb   = (u16*)  (ws + wsl::KTB);
    float* ctx   = (float*)(ws + wsl::CTX);
    u16*   feats = (u16*)  (ws + wsl::FEATS);
    u16*   xb    = (u16*)  (ws + wsl::XB);
    float* logit = (float*)(ws + wsl::LOGIT);
    u16*   wit0  = (u16*)(ws + wsl::WIT0); u16* wht0 = (u16*)(ws + wsl::WHT0);
    u16*   wit1  = (u16*)(ws + wsl::WIT1); u16* wht1 = (u16*)(ws + wsl::WHT1);
    u16*   wqb   = (u16*)(ws + wsl::WQB);  u16* wkb  = (u16*)(ws + wsl::WKB);
    u16*   wvb   = (u16*)(ws + wsl::WVB);  u16* w1b  = (u16*)(ws + wsl::W1B);
    u16*   w2p   = (u16*)(ws + wsl::W2P);
    float* bs0   = (float*)(ws + wsl::BS0);
    float* bs1   = (float*)(ws + wsl::BS1);
    float* b2p   = (float*)(ws + wsl::B2P);
    float* gbuf  = (float*)(ws + wsl::GBUF);
    u16*   hprev = (u16*)  (ws + wsl::HPREV);
    float* cst   = (float*)(ws + wsl::CST);
    unsigned* cnt = (unsigned*)(ws + wsl::CNT);
    int*   meta  = (int*)  (ws + wsl::META);

    const long OUT_OFF  = 0;
    const long ATTN_OFF = (long)BB * TT * NC;                    //   385,600
    const long WORD_OFF = ATTN_OFF + (long)BB * NHEAD * TT * TE; // 8,282,688
    long words_count = (long)out_size - WORD_OFF;
    if (words_count < 0) words_count = 0;
    float* outp = (float*)d_out;

    // --- prep ---------------------------------------------------------------
    zero_u32_k<<<1, 32, 0, stream>>>(cnt, 2);
    transpose_cast_k<<<1024, 256, 0, stream>>>(wit0, wih0, G4, HH);
    transpose_cast_k<<<1024, 256, 0, stream>>>(wht0, whh0, G4, HH);
    transpose_cast_k<<<1024, 256, 0, stream>>>(wit1, wih1, G4, HH);
    transpose_cast_k<<<1024, 256, 0, stream>>>(wht1, whh1, G4, HH);
    cast_bf16_k<<<1024, 256, 0, stream>>>(wqb, wq, (long)HH * HH);
    cast_bf16_k<<<1024, 256, 0, stream>>>(wkb, wk, (long)HH * HH);
    cast_bf16_k<<<1024, 256, 0, stream>>>(wvb, wv, (long)HH * HH);
    cast_bf16_k<<<1024, 256, 0, stream>>>(w1b, w1, (long)2 * HH * HH);
    cast_bf16_k<<<1024, 256, 0, stream>>>(encb, enc, (long)BB * TE * HH);
    pad_w2_k<<<512, 256, 0, stream>>>(w2p, w2);
    pad_b2_k<<<1, 128, 0, stream>>>(b2p, b2);
    bias_sum_k<<<16, 256, 0, stream>>>(bs0, bih0, bhh0, G4);
    bias_sum_k<<<16, 256, 0, stream>>>(bs1, bih1, bhh1, G4);

    // --- embedding ----------------------------------------------------------
    embed_k<<<1024, 256, 0, stream>>>(embf, embb, tok, emb);

    // --- LSTM layer 0 -------------------------------------------------------
    gemm_bf16_k<<<dim3(8, 256), 256, A_TILE_BYTES, stream>>>(
        embb, HH, wit0, G4, bs0, xg, nullptr, BB * TP, G4, HH, 0);
    lstm_scan_k<<<8, 256, A_TILE_BYTES, stream>>>(xg, wht0, nullptr, h0b,
                                                  hprev, cst, gbuf, cnt + 0);

    // --- LSTM layer 1 -------------------------------------------------------
    gemm_bf16_k<<<dim3(8, 256), 256, A_TILE_BYTES, stream>>>(
        h0b, HH, wit1, G4, bs1, xg, nullptr, BB * TP, G4, HH, 0);
    lstm_scan_k<<<8, 256, A_TILE_BYTES, stream>>>(xg, wht1, h1f, h1b,
                                                  hprev, cst, gbuf, cnt + 1);

    // --- QKV projections ----------------------------------------------------
    gemm_bf16_k<<<dim3(2, 256), 256, A_TILE_BYTES, stream>>>(
        h1b, HH, wqb, HH, bq, nullptr, qb, BB * TP, HH, HH, 0);
    gemm_bf16_k<<<dim3(2, 512), 256, A_TILE_BYTES, stream>>>(
        encb, HH, wkb, HH, bk, nullptr, kpb, BB * TE, HH, HH, 0);
    gemm_bf16_k<<<dim3(2, 512), 256, A_TILE_BYTES, stream>>>(
        encb, HH, wvb, HH, bv, nullptr, vbuf, BB * TE, HH, HH, 0);
    kT_pack_k<<<1024, 256, 0, stream>>>(ktb, kpb);

    // --- attention ----------------------------------------------------------
    attn_k<<<dim3(16, NHEAD, BB), 256, 0, stream>>>(qb, ktb, vbuf,
                                                    outp + ATTN_OFF, ctx);

    // --- output MLP ---------------------------------------------------------
    feats_k<<<1024, 256, 0, stream>>>(feats, h1f, ctx);
    gemm_bf16_k<<<dim3(2, 256), 256, A_TILE_BYTES, stream>>>(
        feats, 2 * HH, w1b, HH, b1, nullptr, xb, BB * TP, HH, 2 * HH, 1);
    gemm_bf16_k<<<dim3(1, 256), 256, A_TILE_BYTES, stream>>>(
        xb, HH, w2p, NCP, b2p, logit, nullptr, BB * TP, NCP, HH, 0);
    logsoftmax_k<<<482, 256, 0, stream>>>(outp + OUT_OFF, logit);

    // --- word split + gather ------------------------------------------------
    words_meta_k<<<1, 1, 0, stream>>>(meta, tok);
    words_gather_k<<<1024, 256, 0, stream>>>(outp + WORD_OFF, meta, embf, ctx,
                                             words_count);
}
